// SCD_Loss_5214090297993
// MI455X (gfx1250) — matrix operand order repaired and checked by reading, not compiled, not run on hardware
//
#include <hip/hip_runtime.h>
#include <hip/hip_bf16.h>

typedef __bf16 bf16_t;
typedef __bf16 v8bf  __attribute__((ext_vector_type(8)));
typedef __bf16 v16bf __attribute__((ext_vector_type(16)));
typedef float  v8f   __attribute__((ext_vector_type(8)));
typedef unsigned u32x4 __attribute__((ext_vector_type(4)));
typedef int      i32x4 __attribute__((ext_vector_type(4)));
typedef int      i32x8 __attribute__((ext_vector_type(8)));

#define N_ROWS 4096
#define D_EMB  768
#define D_PROJ 2048
#define LAMBD  0.0051f
#define TEMP_  0.1f
#define ALPHA_ (-1.0f)
#define BETA_  0.005f

#define TILE_M 128
#define TILE_N 128
#define TILE_K 32
#define LDS_S  40   // bf16 pitch: 32 data + 8 pad = 80B rows (TDM pad: 16dw interval, 4dw amount)

// --------------------------------------------------- CDNA5 async copy helpers
__device__ __forceinline__ void async_copy16(unsigned lds_addr, const void* gaddr) {
    // LDS[lds_addr] = MEM[gaddr] (16 bytes), tracked by ASYNCcnt
    asm volatile("global_load_async_to_lds_b128 %0, %1, off"
                 :: "v"(lds_addr), "v"(gaddr) : "memory");
}
__device__ __forceinline__ void wait_async0() {
    asm volatile("s_wait_asynccnt 0x0" ::: "memory");
}
__device__ __forceinline__ unsigned lds_off(const void* p) {
    // generic shared pointer: low 32 bits are the LDS byte address (ISA 10.2 aperture rule)
    return (unsigned)(size_t)p;
}

// ---------------------------------------------- Tensor Data Mover tile loader
// Loads a 128-row x 32-element bf16 tile (row stride = stride_elems in global)
// into LDS with 80-byte row pitch (pad 4 dwords after every 16 dwords).
__device__ __forceinline__ void tdm_tile_128x32(const bf16_t* g, int stride_elems, unsigned lds) {
    unsigned long long ga = (unsigned long long)(size_t)g;
    u32x4 g0;
    g0[0] = 1u;                                                // count=1, user descriptor
    g0[1] = lds;                                               // lds_addr
    g0[2] = (unsigned)ga;                                      // global_addr[31:0]
    g0[3] = (unsigned)((ga >> 32) & 0x01FFFFFFu) | (2u << 30); // global_addr[56:32] | type=2
    i32x8 g1;
    // data_size=2B | pad_enable | pad_interval=16dw(code 3) | pad_amount=4dw(code 3)
    g1[0] = (int)((1u << 16) | (1u << 20) | (3u << 22) | (3u << 25));
    unsigned td0 = (unsigned)stride_elems;                     // tensor_dim0 (>= tile extent)
    unsigned td1 = 1u << 20;                                   // tensor_dim1 (large, no OOB)
    g1[1] = (int)((td0 & 0xFFFFu) << 16);                      // [31:16]=tensor_dim0 lo
    g1[2] = (int)(((td0 >> 16) & 0xFFFFu) | ((td1 & 0xFFFFu) << 16));
    g1[3] = (int)(((td1 >> 16) & 0xFFFFu) | (32u << 16));      // tile_dim0 = 32
    g1[4] = 128;                                               // tile_dim1 = 128
    g1[5] = (int)stride_elems;                                 // tensor_dim0_stride lo
    g1[6] = 0;                                                 // stride hi | tensor_dim1_stride lo
    g1[7] = 0;
    i32x4 z4 = {0, 0, 0, 0};
    i32x8 z8 = {0, 0, 0, 0, 0, 0, 0, 0};
    // 6-arg form (amdgpu-toolchain / clang-23): (g0, g1, g2, g3, g4, cpol)
    __builtin_amdgcn_tensor_load_to_lds(g0, g1, z4, z4, z8, 0);
}

// ---------------------------------------------------------------- utilities
__global__ void cvt_bf16_kernel(const float* __restrict__ in, bf16_t* __restrict__ out, int n) {
    int i = blockIdx.x * blockDim.x + threadIdx.x;
    if (i < n) out[i] = (bf16_t)in[i];
}

// f32 in[R][C] -> bf16 out[C][R] (coalesced both sides via LDS tile)
__global__ __launch_bounds__(256)
void transpose_cvt_kernel(const float* __restrict__ in, bf16_t* __restrict__ out, int R, int C) {
    __shared__ float tile[64][65];
    const int c0 = blockIdx.x * 64, r0 = blockIdx.y * 64;
    const int tx = threadIdx.x & 63, ty = threadIdx.x >> 6;    // 4 rows per pass
    #pragma unroll
    for (int i = 0; i < 16; ++i)
        tile[ty + i * 4][tx] = in[(size_t)(r0 + ty + i * 4) * C + c0 + tx];
    __syncthreads();
    #pragma unroll
    for (int i = 0; i < 16; ++i)
        out[(size_t)(c0 + ty + i * 4) * R + r0 + tx] = (bf16_t)tile[tx][ty + i * 4];
}

// final BN (no affine) fused with transpose: Y[R][D] -> out[D][R] bf16
__global__ __launch_bounds__(256)
void bn_transpose_kernel(const float* __restrict__ Y, const float* __restrict__ mu,
                         const float* __restrict__ rstd, bf16_t* __restrict__ out,
                         int D, int R) {
    __shared__ float tile[64][65];
    const int c0 = blockIdx.x * 64, r0 = blockIdx.y * 64;
    const int tx = threadIdx.x & 63, ty = threadIdx.x >> 6;
    #pragma unroll
    for (int i = 0; i < 16; ++i) {
        int c = c0 + tx;
        tile[ty + i * 4][tx] = (Y[(size_t)(r0 + ty + i * 4) * D + c] - mu[c]) * rstd[c];
    }
    __syncthreads();
    #pragma unroll
    for (int i = 0; i < 16; ++i)
        out[(size_t)(c0 + ty + i * 4) * R + r0 + tx] = (bf16_t)tile[tx][ty + i * 4];
}

__global__ void zero_acc_kernel(float* acc) {
    if (threadIdx.x < 4) acc[threadIdx.x] = 0.f;
}

// ------------------------------------------------------- batchnorm statistics
__global__ void colstats_partial_kernel(const float* __restrict__ Y, float* __restrict__ ps,
                                        float* __restrict__ pss, int rows_per_chunk, int D) {
    int col = blockIdx.x * blockDim.x + threadIdx.x;
    int chunk = blockIdx.y;
    int r0 = chunk * rows_per_chunk;
    float s = 0.f, ss = 0.f;
    for (int r = 0; r < rows_per_chunk; ++r) {
        float v = Y[(size_t)(r0 + r) * D + col];
        s += v; ss += v * v;
    }
    ps [(size_t)chunk * D + col] = s;
    pss[(size_t)chunk * D + col] = ss;
}

__global__ void colstats_final_kernel(const float* __restrict__ ps, const float* __restrict__ pss,
                                      float* __restrict__ mu, float* __restrict__ rstd,
                                      int D, int chunks, float invN) {
    int col = blockIdx.x * blockDim.x + threadIdx.x;
    float s = 0.f, ss = 0.f;
    for (int c = 0; c < chunks; ++c) {
        s  += ps [(size_t)c * D + col];
        ss += pss[(size_t)c * D + col];
    }
    float m = s * invN;
    float var = ss * invN - m * m;       // biased variance (training-mode BN)
    mu[col] = m;
    rstd[col] = rsqrtf(var + 1e-5f);
}

template<bool AFFINE, bool RELU>
__global__ void bn_apply_kernel(const float* __restrict__ X, const float* __restrict__ mu,
                                const float* __restrict__ rstd, const float* __restrict__ g,
                                const float* __restrict__ be, bf16_t* __restrict__ out,
                                int D, int total) {
    int i = blockIdx.x * blockDim.x + threadIdx.x;
    if (i >= total) return;
    int c = i & (D - 1);                 // D = 2048 (power of two)
    float v = (X[i] - mu[c]) * rstd[c];
    if (AFFINE) v = v * g[c] + be[c];
    if (RELU)   v = fmaxf(v, 0.f);
    out[i] = (bf16_t)v;
}

// ------------------------------------------------------------- WMMA GEMM
// C[M,N] (f32) = A(bf16)[M][K] @ Bt(bf16)[N][K]^T  (both operands K-contiguous).
// EPI: 0 = store, 1 = store + bias[n], 2 = Barlow-Twins decorrelation reduction.
// Double-buffered LDS; steady-state tiles staged by the Tensor Data Mover
// (tensor_load_to_lds, TENSORcnt); prologue via global_load_async_to_lds_b128.
template<int EPI>
__global__ __launch_bounds__(256)
void gemm_wmma_kernel(const bf16_t* __restrict__ A, const bf16_t* __restrict__ Bt,
                      float* __restrict__ C, const float* __restrict__ bias,
                      float* __restrict__ acc,
                      int M, int N, int K, int lda, int ldb, int ldc) {
    __shared__ __align__(16) bf16_t sA[2][TILE_M * LDS_S];   // [m][k]
    __shared__ __align__(16) bf16_t sB[2][TILE_N * LDS_S];   // [n][k]
    __shared__ float red[256];

    const int tid  = threadIdx.x;
    const int lane = tid & 31;
    const int wave = tid >> 5;
    const int wm   = wave & 3;     // 4 waves along M -> 32 rows each
    const int wn   = wave >> 2;    // 2 waves along N -> 64 cols each
    const int half = lane >> 4;
    const int lr   = lane & 15;

    const int m0 = blockIdx.y * TILE_M;
    const int n0 = blockIdx.x * TILE_N;

    const bf16_t* Abase = A  + (size_t)m0 * lda;
    const bf16_t* Bbase = Bt + (size_t)n0 * ldb;

    // ---- prologue: per-lane async DMA of tile 0 into buffer 0 ----
    {
        const int kb0 = ((tid * 2 + 0) & 3) * 8;
        const int kb1 = ((tid * 2 + 1) & 3) * 8;
        const int r0c = (tid * 2 + 0) >> 2, r1c = (tid * 2 + 1) >> 2;
        async_copy16(lds_off(&sA[0][r0c * LDS_S + kb0]), (const void*)(Abase + (size_t)r0c * lda + kb0));
        async_copy16(lds_off(&sA[0][r1c * LDS_S + kb1]), (const void*)(Abase + (size_t)r1c * lda + kb1));
        async_copy16(lds_off(&sB[0][r0c * LDS_S + kb0]), (const void*)(Bbase + (size_t)r0c * ldb + kb0));
        async_copy16(lds_off(&sB[0][r1c * LDS_S + kb1]), (const void*)(Bbase + (size_t)r1c * ldb + kb1));
    }
    wait_async0();
    __syncthreads();

    v8f accv[2][4] = {};

    int cur = 0;
    for (int k0 = 0; k0 < K; k0 += TILE_K) {
        const bool has_next = (k0 + TILE_K) < K;
        const int nxt = cur ^ 1;

        // kick the Tensor Data Mover for the next tiles (wave 0 only; EXEC-independent DMA)
        if (has_next && wave == 0) {
            tdm_tile_128x32(Abase + k0 + TILE_K, lda, lds_off(&sA[nxt][0]));
            tdm_tile_128x32(Bbase + k0 + TILE_K, ldb, lds_off(&sB[nxt][0]));
        }

        // ---- fragments from current buffer ----
        // A frag (16x32 bf16): lane (half,lr) holds row lr, K = {half*8..+7} U {16+half*8..+7}
        v16bf afr[2];
        #pragma unroll
        for (int mi = 0; mi < 2; ++mi) {
            const bf16_t* p = &sA[cur][(wm * 32 + mi * 16 + lr) * LDS_S + half * 8];
            v8bf lo = *(const v8bf*)p;
            v8bf hi = *(const v8bf*)(p + 16);
            afr[mi] = __builtin_shufflevector(lo, hi, 0,1,2,3,4,5,6,7,8,9,10,11,12,13,14,15);
        }
        // B frag (32x16 bf16): lane (half,lr) holds col lr, K = {half*8..+7} U {16+half*8..+7}
        v16bf bfr[4];
        #pragma unroll
        for (int ni = 0; ni < 4; ++ni) {
            const bf16_t* p = &sB[cur][(wn * 64 + ni * 16 + lr) * LDS_S + half * 8];
            v8bf lo = *(const v8bf*)p;
            v8bf hi = *(const v8bf*)(p + 16);
            bfr[ni] = __builtin_shufflevector(lo, hi, 0,1,2,3,4,5,6,7,8,9,10,11,12,13,14,15);
        }
        // ---- 8 WMMAs ----
        #pragma unroll
        for (int mi = 0; mi < 2; ++mi)
            #pragma unroll
            for (int ni = 0; ni < 4; ++ni)
                accv[mi][ni] = __builtin_amdgcn_wmma_f32_16x16x32_bf16(
                    false, afr[mi], false, bfr[ni], (short)0, accv[mi][ni], false, false);

        if (has_next) {
            __builtin_amdgcn_s_wait_tensorcnt(0);   // wave 0's DMA complete -> LDS valid
            __syncthreads();
            cur = nxt;
        }
    }

    // ---- epilogue ----
    const int m_base = m0 + wm * 32;
    const int n_base = n0 + wn * 64;
    if (EPI != 2) {
        #pragma unroll
        for (int mi = 0; mi < 2; ++mi) {
            #pragma unroll
            for (int ni = 0; ni < 4; ++ni) {
                int ncol = n_base + ni * 16 + lr;
                float bv = 0.f;
                if (EPI == 1) bv = bias[ncol];
                int mr = m_base + mi * 16 + half * 8;   // C/D: VGPR j -> M = half*8+j
                #pragma unroll
                for (int j = 0; j < 8; ++j)
                    C[(size_t)(mr + j) * ldc + ncol] = accv[mi][ni][j] + bv;
            }
        }
    } else {
        const float invn = 1.0f / (float)N_ROWS;
        float local = 0.f;
        #pragma unroll
        for (int mi = 0; mi < 2; ++mi) {
            #pragma unroll
            for (int ni = 0; ni < 4; ++ni) {
                int n  = n_base + ni * 16 + lr;
                int mr = m_base + mi * 16 + half * 8;
                #pragma unroll
                for (int j = 0; j < 8; ++j) {
                    float v = accv[mi][ni][j] * invn;
                    float d = v - 1.0f;
                    local += (mr + j == n) ? d * d : LAMBD * v * v;
                }
            }
        }
        __syncthreads();
        red[tid] = local;
        __syncthreads();
        for (int s = 128; s > 0; s >>= 1) {
            if (tid < s) red[tid] += red[tid + s];
            __syncthreads();
        }
        if (tid == 0) atomicAdd(acc, red[0]);
    }
}

// ------------------------------------------------------------ cosine diagonal
__global__ void cosine_rows_kernel(const float* __restrict__ z1, const float* __restrict__ z2,
                                   float* __restrict__ acc, int D, int rows) {
    __shared__ float s[8];
    int wave = threadIdx.x >> 5;
    int lane = threadIdx.x & 31;
    int row  = blockIdx.x * 8 + wave;
    const float* a = z1 + (size_t)row * D;
    const float* b = z2 + (size_t)row * D;
    float dot = 0.f, na = 0.f, nb = 0.f;
    for (int k = lane; k < D; k += 32) {
        float x = a[k], y = b[k];
        dot += x * y; na += x * x; nb += y * y;
    }
    for (int off = 16; off > 0; off >>= 1) {
        dot += __shfl_down(dot, off, 32);
        na  += __shfl_down(na,  off, 32);
        nb  += __shfl_down(nb,  off, 32);
    }
    if (lane == 0) s[wave] = dot / fmaxf(sqrtf(na) * sqrtf(nb), 1e-8f);
    __syncthreads();
    if (threadIdx.x == 0) {
        float t = 0.f;
        for (int i = 0; i < 8; ++i) t += s[i];
        atomicAdd(acc, t);
    }
}

__global__ void finalize_kernel(const float* __restrict__ acc, float* __restrict__ out) {
    // acc[0] = decorrelation (on_diag + lambd*off_diag), acc[1] = sum of per-row cosines
    out[0] = ALPHA_ * (acc[1] / (float)N_ROWS / TEMP_) + BETA_ * acc[0];
}

// ---------------------------------------------------------------- launcher
extern "C" void kernel_launch(void* const* d_in, const int* in_sizes, int n_in,
                              void* d_out, int out_size, void* d_ws, size_t ws_size,
                              hipStream_t stream) {
    (void)in_sizes; (void)n_in; (void)out_size; (void)ws_size;
    const float* z1  = (const float*)d_in[0];
    const float* z2  = (const float*)d_in[1];
    const float* W1  = (const float*)d_in[2];
    const float* b1  = (const float*)d_in[3];
    const float* g1  = (const float*)d_in[4];
    const float* be1 = (const float*)d_in[5];
    const float* W2  = (const float*)d_in[6];
    const float* g2  = (const float*)d_in[7];
    const float* be2 = (const float*)d_in[8];
    const float* W3  = (const float*)d_in[9];

    char* ws = (char*)d_ws;
    bf16_t* W1t  = (bf16_t*)(ws + 0);            // Wt[2048][768]   = 3,145,728 B
    bf16_t* W2t  = (bf16_t*)(ws + 3145728);      // Wt[2048][2048]  = 8,388,608
    bf16_t* W3t  = (bf16_t*)(ws + 11534336);     // 8,388,608
    bf16_t* Zb   = (bf16_t*)(ws + 19922944);     // [4096][768]     = 6,291,456
    float*  Y    = (float*) (ws + 26214400);     // [4096][2048] f32 = 33,554,432
    bf16_t* Hb   = (bf16_t*)(ws + 59768832);     // [4096][2048]    = 16,777,216
    bf16_t* P1t  = (bf16_t*)(ws + 76546048);     // p1^T [2048][4096] = 16,777,216
    bf16_t* P2t  = (bf16_t*)(ws + 93323264);     // p2^T [2048][4096] = 16,777,216
    float*  psum = (float*) (ws + 110100480);    // 32*2048*4 = 262,144
    float*  psq  = (float*) (ws + 110362624);    // 262,144
    float*  mu   = (float*) (ws + 110624768);    // 8,192
    float*  rstd = (float*) (ws + 110632960);    // 8,192
    float*  acc  = (float*) (ws + 110641152);    // scalars

    const int TOT = N_ROWS * D_PROJ;             // 8,388,608

    zero_acc_kernel<<<1, 32, 0, stream>>>(acc);

    // weights: fp32 [K][N] -> bf16 transposed [N][K]  (K-contiguous B operands)
    transpose_cvt_kernel<<<dim3(D_PROJ / 64, D_EMB / 64),  256, 0, stream>>>(W1, W1t, D_EMB,  D_PROJ);
    transpose_cvt_kernel<<<dim3(D_PROJ / 64, D_PROJ / 64), 256, 0, stream>>>(W2, W2t, D_PROJ, D_PROJ);
    transpose_cvt_kernel<<<dim3(D_PROJ / 64, D_PROJ / 64), 256, 0, stream>>>(W3, W3t, D_PROJ, D_PROJ);

    // self-contrast diagonal (independent of projector path)
    cosine_rows_kernel<<<N_ROWS / 8, 256, 0, stream>>>(z1, z2, acc + 1, D_EMB, N_ROWS);

    dim3 blk(256);
    dim3 grid_g1(D_PROJ / TILE_N, N_ROWS / TILE_M);
    dim3 grid_cs(D_PROJ / 256, 32);
    dim3 grid_tp(D_PROJ / 64, N_ROWS / 64);

    const float invN = 1.0f / (float)N_ROWS;

    for (int branch = 0; branch < 2; ++branch) {
        const float* z = branch == 0 ? z1 : z2;
        bf16_t* Pt = branch == 0 ? P1t : P2t;

        cvt_bf16_kernel<<<(N_ROWS * D_EMB + 255) / 256, 256, 0, stream>>>(z, Zb, N_ROWS * D_EMB);

        // layer 1: z @ W1 + b1 -> BN(g1,be1) -> ReLU
        gemm_wmma_kernel<1><<<grid_g1, blk, 0, stream>>>(
            Zb, W1t, Y, b1, nullptr, N_ROWS, D_PROJ, D_EMB, D_EMB, D_EMB, D_PROJ);
        colstats_partial_kernel<<<grid_cs, blk, 0, stream>>>(Y, psum, psq, N_ROWS / 32, D_PROJ);
        colstats_final_kernel<<<D_PROJ / 256, blk, 0, stream>>>(psum, psq, mu, rstd, D_PROJ, 32, invN);
        bn_apply_kernel<true, true><<<TOT / 256, blk, 0, stream>>>(Y, mu, rstd, g1, be1, Hb, D_PROJ, TOT);

        // layer 2: h @ W2 -> BN(g2,be2) -> ReLU
        gemm_wmma_kernel<0><<<grid_g1, blk, 0, stream>>>(
            Hb, W2t, Y, nullptr, nullptr, N_ROWS, D_PROJ, D_PROJ, D_PROJ, D_PROJ, D_PROJ);
        colstats_partial_kernel<<<grid_cs, blk, 0, stream>>>(Y, psum, psq, N_ROWS / 32, D_PROJ);
        colstats_final_kernel<<<D_PROJ / 256, blk, 0, stream>>>(psum, psq, mu, rstd, D_PROJ, 32, invN);
        bn_apply_kernel<true, true><<<TOT / 256, blk, 0, stream>>>(Y, mu, rstd, g2, be2, Hb, D_PROJ, TOT);

        // layer 3: h @ W3 -> BN(no affine) fused with transpose -> p^T (bf16)
        gemm_wmma_kernel<0><<<grid_g1, blk, 0, stream>>>(
            Hb, W3t, Y, nullptr, nullptr, N_ROWS, D_PROJ, D_PROJ, D_PROJ, D_PROJ, D_PROJ);
        colstats_partial_kernel<<<grid_cs, blk, 0, stream>>>(Y, psum, psq, N_ROWS / 32, D_PROJ);
        colstats_final_kernel<<<D_PROJ / 256, blk, 0, stream>>>(psum, psq, mu, rstd, D_PROJ, 32, invN);
        bn_transpose_kernel<<<grid_tp, blk, 0, stream>>>(Y, mu, rstd, Pt, D_PROJ, N_ROWS);
    }

    // c = p1^T @ p2 / N with fused decorrelation reduction
    // A = P1t [2048][4096], Bt = P2t [2048][4096] -> plain K-contiguous GEMM
    dim3 grid_c(D_PROJ / TILE_N, D_PROJ / TILE_M);
    gemm_wmma_kernel<2><<<grid_c, blk, 0, stream>>>(
        P1t, P2t, nullptr, nullptr, acc, D_PROJ, D_PROJ, N_ROWS, N_ROWS, N_ROWS, 0);

    finalize_kernel<<<1, 1, 0, stream>>>(acc, (float*)d_out);
}
